// FastTreeLSTM_6305011990802
// MI455X (gfx1250) — compile-verified, matrix operand order checked by reading, not run on hardware
//
#include <hip/hip_runtime.h>
#include <hip/hip_bf16.h>
#include <math.h>

#define NTOT  131072
#define LVLS  8
#define NPL   16384        // nodes per level
#define HD    128
#define GATES 512          // 4*H
#define KTOT  256          // 128 (x-path) + 128 (h-path)

typedef __attribute__((ext_vector_type(16))) __bf16 v16bf;
typedef __attribute__((ext_vector_type(8)))  float  v8f;

__device__ __forceinline__ unsigned short f2bf(float x) {
    union { float f; unsigned u; } cv; cv.f = x;
    unsigned u = cv.u;
    unsigned r = u + 0x7FFFu + ((u >> 16) & 1u);   // round-to-nearest-even
    return (unsigned short)(r >> 16);
}

__device__ __forceinline__ float sigmoidf_(float x) {
    return 1.0f / (1.0f + __expf(-x));
}

// Map local K (0..31) inside one 32-K bf16 WMMA chunk to (khalf, elem):
// lanes 0-15 (khalf=0) hold K {0..7, 16..23}; lanes 16-31 hold K {8..15, 24..31};
// VGPR v holds two packed K values (elem = 2v / 2v+1).
__device__ __forceinline__ void frag_pos(int klocal, int& khalf, int& e) {
    khalf  = (klocal >> 3) & 1;
    int kb = klocal & 7;
    int hi = klocal >> 4;
    e = 2 * ((kb >> 1) + 4 * hi) + (kb & 1);
}

// ---------------------------------------------------------------- zero h0
__global__ void k_zero(float* __restrict__ p, int n4) {
    int i = blockIdx.x * blockDim.x + threadIdx.x;
    if (i < n4) ((float4*)p)[i] = make_float4(0.f, 0.f, 0.f, 0.f);
}

// ------------------------------------------------- h0 = scatter_add(features)
__global__ void k_scatter(const float* __restrict__ feat,
                          const int* __restrict__ imap,
                          float* __restrict__ h0) {
    int t = blockIdx.x * blockDim.x + threadIdx.x;   // NTOT*HD threads
    int node = t >> 7, col = t & 127;
    if (node < NTOT) {
        int dst = imap[node];
        atomicAdd(&h0[(size_t)dst * HD + col], feat[(size_t)node * HD + col]);
    }
}

// ---------------- pack Wc = [Wx ; Wh] (256x512 f32) into bf16 WMMA B-fragments
// layout: wfrag[(((nt*8)+kc)*32 + lane)*16 + e]
__global__ void k_wconv(const float* __restrict__ wx,
                        const float* __restrict__ wh,
                        unsigned short* __restrict__ wfrag) {
    int id = blockIdx.x * blockDim.x + threadIdx.x;  // KTOT*GATES = 131072
    if (id >= KTOT * GATES) return;
    int e = id & 15, lane = (id >> 4) & 31, kc = (id >> 9) & 7, nt = id >> 12;
    int v = e >> 1, lo = e & 1, khalf = lane >> 4;
    int klocal = ((v < 4) ? 2 * v : 2 * (v - 4) + 16) + 8 * khalf + lo;
    int k    = kc * 32 + klocal;
    int ncol = (lane & 15) + nt * 16;
    float w = (k < HD) ? wx[(size_t)k * GATES + ncol]
                       : wh[(size_t)(k - HD) * GATES + ncol];
    wfrag[id] = f2bf(w);
}

// --------------------------------------------------------- one DAG level step
// block = 256 threads (8 wave32), owns 16 rows of the level.
__global__ void __launch_bounds__(256)
k_level(const float* __restrict__ h0,
        const float* __restrict__ bias,
        const int*   __restrict__ edge_child,
        const unsigned short* __restrict__ wfrag,
        float* __restrict__ hws,
        float* __restrict__ cws,
        int level) {
    __shared__ __align__(32) unsigned short aFrag[8 * 32 * 16]; // [kc][lane][e] bf16, 8KB
    __shared__ __align__(32) float gatesT[GATES * 16];          // [col][row] f32, 32KB
    __shared__ __align__(32) float csumS[16 * HD];              // [row][col] f32, 8KB

    const int rowbase = blockIdx.x * 16;        // local row within level
    const int tid  = threadIdx.x;

    // Stage A = [h0_row(128) || h_sum(128)] as bf16 fragments; also c_sum (f32).
    for (int it = 0; it < 8; ++it) {
        int idx = tid + it * 256;               // 0..2047
        int r   = idx >> 7;                     // 0..15
        int col = idx & 127;                    // 0..127
        int gnode = level * NPL + rowbase + r;
        float a0 = h0[(size_t)gnode * HD + col];
        float hs = 0.f, cs = 0.f;
        if (level > 0) {
            const int* ec = edge_child + (size_t)(level - 1) * NPL * 4
                                       + (size_t)(rowbase + r) * 4;
#pragma unroll
            for (int j = 0; j < 4; ++j) {
                int ch = ec[j];                 // global index into level-1
                hs += hws[(size_t)ch * HD + col];
                cs += cws[(size_t)ch * HD + col];
            }
        }
        csumS[idx] = cs;
        int kh0, e0, kh1, e1;
        frag_pos(col & 31, kh0, e0);            // K = col        -> kc = col>>5
        frag_pos(col & 31, kh1, e1);            // K = 128 + col  -> kc = 4 + (col>>5)
        int kcA = col >> 5;
        aFrag[((kcA      ) * 32 + (r + 16 * kh0)) * 16 + e0] = f2bf(a0);
        aFrag[((kcA + 4  ) * 32 + (r + 16 * kh1)) * 16 + e1] = f2bf(hs);
    }
    __syncthreads();

    const int wave = tid >> 5;
    const int lane = tid & 31;
    const int hi   = lane >> 4;

    // Each wave: 4 N-tiles of 16 gate columns; K=256 -> 8 WMMAs per tile.
#pragma unroll
    for (int t = 0; t < 4; ++t) {
        int nt = wave * 4 + t;
        v8f acc = {};
#pragma unroll
        for (int kc = 0; kc < 8; ++kc) {
            v16bf a = *(const v16bf*)&aFrag[(kc * 32 + lane) * 16];
            v16bf b = *(const v16bf*)&wfrag[(((size_t)nt * 8 + kc) * 32 + lane) * 16];
            acc = __builtin_amdgcn_wmma_f32_16x16x32_bf16(
                false, a, false, b, (short)0, acc, false, false);
        }
        int ncol = nt * 16 + (lane & 15);
        float bb = bias[ncol];
#pragma unroll
        for (int v = 0; v < 8; ++v) acc[v] += bb;
        // D layout: lane holds col = lane&15, rows v + 8*hi -> store transposed,
        // 8 consecutive f32 => one 32-byte LDS store.
        *(v8f*)&gatesT[ncol * 16 + 8 * hi] = acc;
    }
    __syncthreads();

    // Gate nonlinearity: c = sig(i)*tanh(g) + sig(f)*c_sum; h = sig(o)*tanh(c)
    for (int it = 0; it < 8; ++it) {
        int idx = tid + it * 256;
        int r = idx >> 7, j = idx & 127;
        float gi = gatesT[(j        ) * 16 + r];
        float gf = gatesT[(j + 128  ) * 16 + r];
        float gg = gatesT[(j + 256  ) * 16 + r];
        float go = gatesT[(j + 384  ) * 16 + r];
        float cs = csumS[idx];
        float c  = sigmoidf_(gi) * tanhf(gg) + sigmoidf_(gf) * cs;
        float h  = sigmoidf_(go) * tanhf(c);
        size_t g = (size_t)(level * NPL + rowbase + r) * HD + j;
        hws[g] = h;
        cws[g] = c;
    }
}

// ------------------------------------------------------ out = hidden[index_map]
__global__ void k_gather(const float* __restrict__ hws,
                         const int* __restrict__ imap,
                         float* __restrict__ out, int n4) {
    int t = blockIdx.x * blockDim.x + threadIdx.x;   // one float4 per thread
    if (t >= n4) return;
    int node = t >> 5, g = t & 31;
    ((float4*)out)[t] = ((const float4*)&hws[(size_t)imap[node] * HD])[g];
}

extern "C" void kernel_launch(void* const* d_in, const int* in_sizes, int n_in,
                              void* d_out, int out_size, void* d_ws, size_t ws_size,
                              hipStream_t stream) {
    (void)in_sizes; (void)n_in; (void)out_size; (void)ws_size;
    const float* feat = (const float*)d_in[0];
    const float* wx   = (const float*)d_in[1];
    const float* wh   = (const float*)d_in[2];
    const float* bias = (const float*)d_in[3];
    const int*   imap = (const int*)d_in[4];
    const int*   ec   = (const int*)d_in[5];
    float* out = (float*)d_out;

    float* h0  = (float*)d_ws;                       // [N,128] f32
    float* hws = h0  + (size_t)NTOT * HD;            // [N,128] f32
    float* cws = hws + (size_t)NTOT * HD;            // [N,128] f32
    unsigned short* wfrag = (unsigned short*)(cws + (size_t)NTOT * HD); // 256KB bf16

    const int n4 = NTOT * HD / 4;
    k_zero   <<<(n4 + 255) / 256, 256, 0, stream>>>(h0, n4);
    k_scatter<<<(NTOT * HD) / 256, 256, 0, stream>>>(feat, imap, h0);
    k_wconv  <<<(KTOT * GATES) / 256, 256, 0, stream>>>(wx, wh, wfrag);
    for (int l = 0; l < LVLS; ++l)
        k_level<<<NPL / 16, 256, 0, stream>>>(h0, bias, ec, wfrag, hws, cws, l);
    k_gather <<<(n4 + 255) / 256, 256, 0, stream>>>(hws, imap, out, n4);
}
